// MyBlock_64527588655618
// MI455X (gfx1250) — matrix operand order repaired and checked by reading, compiled, not run on hardware
//
#include <hip/hip_runtime.h>
#include <hip/hip_bf16.h>
#include <math.h>

typedef __attribute__((ext_vector_type(16))) __bf16 v16bf;
typedef __attribute__((ext_vector_type(8)))  __bf16 v8bf;
typedef __attribute__((ext_vector_type(8)))  float  v8f;

#define LDSS 40  // LDS row stride in bf16 elems (80B -> 16B-aligned fragment loads)

__device__ __forceinline__ float silu_f(float x) { return x / (1.0f + __expf(-x)); }
__device__ __forceinline__ float softplus_f(float x) {
    return (x > 20.0f) ? x : log1pf(__expf(x));
}

// fp32 (N,K) weights -> bf16 (Npad,K), zero-padded rows. out index == in index.
__global__ __launch_bounds__(256)
void cvt_w_kernel(const float* __restrict__ W, __bf16* __restrict__ out,
                  int total_in) {
    int idx = blockIdx.x * 256 + threadIdx.x;
    out[idx] = (idx < total_in) ? (__bf16)W[idx] : (__bf16)0.0f;
}

// EPI: 0=none, 1=silu, 2=add residual aux[row*ldc+col], 3=softplus(acc+aux[col])
// A: fp32 (M x K, lda), W: bf16 (Npad x K, ldw) pre-padded -> no bounds check.
// Double-buffered LDS: stage tile k+1 (async bf16 W copy + A cvt) while WMMAs
// consume tile k; one s_wait_asynccnt + barrier per K-step.
template <int EPI>
__global__ __launch_bounds__(256)
void gemm_bf16_wmma(const float* __restrict__ A, const __bf16* __restrict__ W,
                    float* __restrict__ C, int M, int N, int K,
                    int lda, int ldw, int ldc, const float* __restrict__ aux) {
    __shared__ __bf16 As[2][128 * LDSS];
    __shared__ __bf16 Bs[2][128 * LDSS];

    const int t    = threadIdx.x;
    const int lane = t & 31;
    const int wid  = t >> 5;
    const int wm   = wid & 3;      // 4 M-slices of 32 rows
    const int wn   = wid >> 2;     // 2 N-slices of 64 cols
    const int lr   = lane & 15;
    const int hs   = lane >> 4;    // half-select (lanes 16-31)
    const int m0   = blockIdx.y * 128;
    const int n0   = blockIdx.x * 128;

    auto stage = [&](int buf, int k0) {
        // W tile: async bf16 copy global -> LDS (512 x 16B, 2 per thread)
#pragma unroll
        for (int i = 0; i < 2; ++i) {
            int slot = t + i * 256;
            int row  = slot >> 2;            // 0..127
            int kc   = (slot & 3) << 3;      // 0,8,16,24
            unsigned ldsa = (unsigned)(uintptr_t)(&Bs[buf][row * LDSS + kc]);
            const __bf16* g = W + (size_t)(n0 + row) * ldw + (k0 + kc);
            asm volatile("global_load_async_to_lds_b128 %0, %1, off"
                         :: "v"(ldsa), "v"(g) : "memory");
        }
        // A tile: fp32 load + cvt -> LDS bf16
#pragma unroll
        for (int i = 0; i < 4; ++i) {
            int slot = t + i * 256;          // 1024 slots of 4 floats
            int row  = slot >> 3;
            int kc   = (slot & 7) << 2;
            float4 av = *(const float4*)(A + (size_t)(m0 + row) * lda + (k0 + kc));
            __bf16* ap = &As[buf][row * LDSS + kc];
            ap[0] = (__bf16)av.x; ap[1] = (__bf16)av.y;
            ap[2] = (__bf16)av.z; ap[3] = (__bf16)av.w;
        }
    };

    v8f acc[2][4];
#pragma unroll
    for (int mi = 0; mi < 2; ++mi)
#pragma unroll
        for (int ni = 0; ni < 4; ++ni) acc[mi][ni] = (v8f)0.0f;

    stage(0, 0);
    asm volatile("s_wait_asynccnt 0x0" ::: "memory");
    __syncthreads();

    int buf = 0;
    for (int k0 = 0; k0 < K; k0 += 32) {
        if (k0 + 32 < K) stage(buf ^ 1, k0 + 32);   // overlap with WMMAs below

        // A fragment (16x32): lane lr = row M; elems 0-7 -> K=hs*8+0..7, 8-15 -> +16
        v16bf afrag[2];
#pragma unroll
        for (int mi = 0; mi < 2; ++mi) {
            const __bf16* ap = &As[buf][(wm * 32 + mi * 16 + lr) * LDSS + hs * 8];
            v8bf a0 = *(const v8bf*)ap;
            v8bf a1 = *(const v8bf*)(ap + 16);
            afrag[mi] = __builtin_shufflevector(a0, a1, 0, 1, 2, 3, 4, 5, 6, 7,
                                                8, 9, 10, 11, 12, 13, 14, 15);
        }
        // B fragment (32x16): lane lr = col N; elems 0-7 -> K=hs*8+0..7, 8-15 -> +16
        v16bf bfrag[4];
#pragma unroll
        for (int ni = 0; ni < 4; ++ni) {
            const __bf16* bp = &Bs[buf][(wn * 64 + ni * 16 + lr) * LDSS + hs * 8];
            v8bf b0 = *(const v8bf*)bp;
            v8bf b1 = *(const v8bf*)(bp + 16);
            bfrag[ni] = __builtin_shufflevector(b0, b1, 0, 1, 2, 3, 4, 5, 6, 7,
                                                8, 9, 10, 11, 12, 13, 14, 15);
        }
#pragma unroll
        for (int mi = 0; mi < 2; ++mi)
#pragma unroll
            for (int ni = 0; ni < 4; ++ni)
                acc[mi][ni] = __builtin_amdgcn_wmma_f32_16x16x32_bf16(
                    false, afrag[mi], false, bfrag[ni],
                    (short)0, acc[mi][ni], false, false);

        asm volatile("s_wait_asynccnt 0x0" ::: "memory");
        __syncthreads();            // publish next buffer to all waves
        buf ^= 1;
    }

    // Epilogue + store. D layout: VGPR r -> row r + 8*hs; lane lr -> col.
    const bool fullN = (n0 + 128 <= N);
#pragma unroll
    for (int mi = 0; mi < 2; ++mi) {
#pragma unroll
        for (int ni = 0; ni < 4; ++ni) {
            int col = n0 + wn * 64 + ni * 16 + lr;
            if (!fullN && col >= N) continue;
            size_t base = (size_t)(m0 + wm * 32 + mi * 16 + hs * 8) * ldc + col;
#pragma unroll
            for (int r = 0; r < 8; ++r) {
                float v = acc[mi][ni][r];
                if (EPI == 1) v = silu_f(v);
                else if (EPI == 2) v += aux[base + (size_t)r * ldc];
                else if (EPI == 3) v = softplus_f(v + aux[col]);
                C[base + (size_t)r * ldc] = v;
            }
        }
    }
}

// LayerNorm (no bias, scale only), D must be 1024; one block per row.
__global__ __launch_bounds__(256)
void ln_kernel(const float* __restrict__ x, const float* __restrict__ w,
               float* __restrict__ out, int D) {
    const int row = blockIdx.x;
    const int t   = threadIdx.x;
    const float* xr = x + (size_t)row * D;
    float v[4], s = 0.f, s2 = 0.f;
#pragma unroll
    for (int i = 0; i < 4; ++i) {
        v[i] = xr[t + i * 256];
        s += v[i];
        s2 += v[i] * v[i];
    }
    __shared__ float r1[256], r2[256];
    r1[t] = s; r2[t] = s2;
    __syncthreads();
    for (int off = 128; off > 0; off >>= 1) {
        if (t < off) { r1[t] += r1[t + off]; r2[t] += r2[t + off]; }
        __syncthreads();
    }
    float mu  = r1[0] / (float)D;
    float var = r2[0] / (float)D - mu * mu;
    float inv = rsqrtf(var + 1e-5f);
#pragma unroll
    for (int i = 0; i < 4; ++i) {
        int c = t + i * 256;
        out[(size_t)row * D + c] = (v[i] - mu) * inv * w[c];
    }
}

// Causal depthwise conv (K=4) + bias + SiLU. xz is (B*L, 4096); xm = cols [0,2048).
__global__ __launch_bounds__(256)
void conv_silu_kernel(const float* __restrict__ xz, const float* __restrict__ cw,
                      const float* __restrict__ cb, float* __restrict__ xs,
                      int total) {
    int idx = blockIdx.x * 256 + threadIdx.x;
    if (idx >= total) return;
    int d = idx & 2047;
    int m = idx >> 11;      // token = b*L + l
    int l = m & 2047;
    int b = m >> 11;
    float s = cb[d];
#pragma unroll
    for (int k = 0; k < 4; ++k) {
        int ll = l + k - 3;
        if (ll >= 0)
            s += cw[d * 4 + k] * xz[(size_t)((b << 11) | ll) * 4096 + d];
    }
    xs[idx] = silu_f(s);
}

// Selective scan: one thread per (b, channel d); 16-state recurrence over L.
__global__ __launch_bounds__(256)
void scan_kernel(const float* __restrict__ dbl, const float* __restrict__ dt,
                 const float* __restrict__ xs, const float* __restrict__ A_log,
                 float* __restrict__ y, int L) {
    int tid = blockIdx.x * 256 + threadIdx.x;   // 0..4095
    int d = tid & 2047;
    int b = tid >> 11;
    float Aa[16], h[16];
#pragma unroll
    for (int n = 0; n < 16; ++n) {
        Aa[n] = -__expf(A_log[d * 16 + n]);
        h[n] = 0.f;
    }
    for (int l = 0; l < L; ++l) {
        size_t tok = (size_t)b * L + l;
        float dtv = dt[tok * 2048 + d];
        float xv  = xs[tok * 2048 + d];
        const float* bc = dbl + tok * 96;
        float yv = 0.f;
#pragma unroll
        for (int n = 0; n < 16; ++n) {
            float dA = __expf(dtv * Aa[n]);
            h[n] = h[n] * dA + dtv * bc[64 + n] * xv;
            yv += h[n] * bc[80 + n];
        }
        y[tok * 2048 + d] = yv;
    }
}

// y = (y + xs*D) * silu(z);  z = xz[:, 2048+d]
__global__ __launch_bounds__(256)
void gate_kernel(float* __restrict__ y, const float* __restrict__ xs,
                 const float* __restrict__ Dv, const float* __restrict__ xz,
                 int total) {
    int idx = blockIdx.x * 256 + threadIdx.x;
    if (idx >= total) return;
    int d = idx & 2047;
    int m = idx >> 11;
    float zv = xz[(size_t)m * 4096 + 2048 + d];
    float v  = y[idx] + xs[idx] * Dv[d];
    y[idx] = v * silu_f(zv);
}

extern "C" void kernel_launch(void* const* d_in, const int* in_sizes, int n_in,
                              void* d_out, int out_size, void* d_ws, size_t ws_size,
                              hipStream_t stream) {
    const float* x         = (const float*)d_in[0];   // (2,2048,1024)
    const float* ln1_w     = (const float*)d_in[1];
    const float* in_proj_w = (const float*)d_in[2];   // (4096,1024)
    const float* conv_w    = (const float*)d_in[3];   // (2048,1,4)
    const float* conv_b    = (const float*)d_in[4];
    const float* x_proj_w  = (const float*)d_in[5];   // (96,2048)
    const float* dt_proj_w = (const float*)d_in[6];   // (2048,64)
    const float* dt_proj_b = (const float*)d_in[7];
    const float* A_log     = (const float*)d_in[8];   // (2048,16)
    const float* Dvec      = (const float*)d_in[9];
    const float* out_proj_w= (const float*)d_in[10];  // (1024,2048)
    const float* ln2_w     = (const float*)d_in[11];
    const float* fc_w      = (const float*)d_in[12];  // (4096,1024)
    const float* proj_w    = (const float*)d_in[13];  // (1024,4096)
    float* out = (float*)d_out;

    const int M = 2 * 2048;   // B*L tokens
    float* ws = (float*)d_ws;
    size_t oXZ  = 0;
    size_t oH   = oXZ  + (size_t)M * 4096;   // ln scratch (reused for ln2)
    size_t oXS  = oH   + (size_t)M * 1024;
    size_t oDBL = oXS  + (size_t)M * 2048;
    size_t oDT  = oDBL + (size_t)M * 96;
    size_t oY   = oDT  + (size_t)M * 2048;
    size_t oX1  = oY   + (size_t)M * 2048;
    size_t oWB  = oX1  + (size_t)M * 1024;   // bf16 weight pool starts here
    float* xz  = ws + oXZ;
    float* h   = ws + oH;
    float* xs  = ws + oXS;
    float* dbl = ws + oDBL;
    float* dtb = ws + oDT;
    float* y   = ws + oY;
    float* x1  = ws + oX1;
    float* ffn = ws + oXZ;   // reuse xz after gating

    // bf16 weight pool (rows padded up to multiple of 128, zero-filled)
    __bf16* wb      = (__bf16*)(ws + oWB);
    __bf16* wb_in   = wb;                       // 4096 x 1024
    __bf16* wb_xp   = wb_in  + 4096 * 1024;     //  128 x 2048 (96 padded)
    __bf16* wb_dt   = wb_xp  + 128  * 2048;     // 2048 x 64
    __bf16* wb_out  = wb_dt  + 2048 * 64;       // 1024 x 2048
    __bf16* wb_fc   = wb_out + 1024 * 2048;     // 4096 x 1024
    __bf16* wb_pr   = wb_fc  + 4096 * 1024;     // 1024 x 4096

    dim3 blk(256);

    // 0) one-shot fp32 -> bf16 weight conversion (padded)
    cvt_w_kernel<<<(4096 * 1024) / 256, blk, 0, stream>>>(in_proj_w, wb_in, 4096 * 1024);
    cvt_w_kernel<<<(128 * 2048) / 256, blk, 0, stream>>>(x_proj_w, wb_xp, 96 * 2048);
    cvt_w_kernel<<<(2048 * 64) / 256, blk, 0, stream>>>(dt_proj_w, wb_dt, 2048 * 64);
    cvt_w_kernel<<<(1024 * 2048) / 256, blk, 0, stream>>>(out_proj_w, wb_out, 1024 * 2048);
    cvt_w_kernel<<<(4096 * 1024) / 256, blk, 0, stream>>>(fc_w, wb_fc, 4096 * 1024);
    cvt_w_kernel<<<(1024 * 4096) / 256, blk, 0, stream>>>(proj_w, wb_pr, 1024 * 4096);

    // 1) LN1
    ln_kernel<<<M, blk, 0, stream>>>(x, ln1_w, h, 1024);
    // 2) xz = h @ in_proj_w.T   (M x 4096, K=1024)
    gemm_bf16_wmma<0><<<dim3(4096 / 128, M / 128), blk, 0, stream>>>(
        h, wb_in, xz, M, 4096, 1024, 1024, 1024, 4096, nullptr);
    // 3) conv + silu -> xs (M x 2048)
    conv_silu_kernel<<<(M * 2048) / 256, blk, 0, stream>>>(xz, conv_w, conv_b, xs,
                                                           M * 2048);
    // 4) dbl = xs @ x_proj_w.T  (M x 96, K=2048)
    gemm_bf16_wmma<0><<<dim3(1, M / 128), blk, 0, stream>>>(
        xs, wb_xp, dbl, M, 96, 2048, 2048, 2048, 96, nullptr);
    // 5) dt = softplus(dbl[:, :64] @ dt_proj_w.T + b)  (M x 2048, K=64, lda=96)
    gemm_bf16_wmma<3><<<dim3(2048 / 128, M / 128), blk, 0, stream>>>(
        dbl, wb_dt, dtb, M, 2048, 64, 96, 64, 2048, dt_proj_b);
    // 6) selective scan -> y
    scan_kernel<<<4096 / 256, blk, 0, stream>>>(dbl, dtb, xs, A_log, y, 2048);
    // 7) y = (y + xs*D) * silu(z)
    gate_kernel<<<(M * 2048) / 256, blk, 0, stream>>>(y, xs, Dvec, xz, M * 2048);
    // 8) x1 = x + y @ out_proj_w.T  (M x 1024, K=2048)
    gemm_bf16_wmma<2><<<dim3(1024 / 128, M / 128), blk, 0, stream>>>(
        y, wb_out, x1, M, 1024, 2048, 2048, 2048, 1024, x);
    // 9) LN2
    ln_kernel<<<M, blk, 0, stream>>>(x1, ln2_w, h, 1024);
    // 10) ffn = silu(h @ fc_w.T)  (M x 4096, K=1024)
    gemm_bf16_wmma<1><<<dim3(4096 / 128, M / 128), blk, 0, stream>>>(
        h, wb_fc, ffn, M, 4096, 1024, 1024, 1024, 4096, nullptr);
    // 11) out = x1 + ffn @ proj_w.T  (M x 1024, K=4096)
    gemm_bf16_wmma<2><<<dim3(1024 / 128, M / 128), blk, 0, stream>>>(
        ffn, wb_pr, out, M, 1024, 4096, 4096, 4096, 1024, x1);
}
